// LSTMAutoEncoderWithAttention_85804856639907
// MI455X (gfx1250) — matrix operand order repaired and checked by reading, not compiled, not run on hardware
//
#include <hip/hip_runtime.h>

// ---------------------------------------------------------------------------
// Types for gfx1250 WMMA (wave32): A/B = 16 x bf16 per lane, C/D = 8 x f32.
// ---------------------------------------------------------------------------
typedef __bf16 bf16_t;
typedef __bf16 v16bf __attribute__((ext_vector_type(16)));
typedef float  v8f   __attribute__((ext_vector_type(8)));

union FragBF {
    v16bf        v;
    unsigned int u[8];
    bf16_t       h[16];
};

// ---------------------------------------------------------------------------
// CDNA5 async memory->LDS copy (ASYNCcnt-tracked, no VGPR round trip).
// ---------------------------------------------------------------------------
__device__ __forceinline__ void async_ld_b128(unsigned int lds_byte_addr,
                                              const void* gsrc)
{
    asm volatile("global_load_async_to_lds_b128 %0, %1, off"
                 :: "v"(lds_byte_addr), "v"(gsrc)
                 : "memory");
}

__device__ __forceinline__ void wait_asynccnt0()
{
    asm volatile("s_wait_asynccnt 0x0" ::: "memory");
}

__device__ __forceinline__ unsigned int lds_addr(const void* p)
{
    return (unsigned int)(uintptr_t)p;   // low 32 bits = LDS offset
}

// ---------------------------------------------------------------------------
// Tiled WMMA GEMM, double-buffered async LDS staging, configurable tiling.
//   BMODE = 0 : out[M,N] = A[M,K](bf16,rm) * W[N,K]^T (bf16,rm)
//   BMODE = 1 : out[M,N] = A[M,K](bf16,rm) * Bk[K,N]  (bf16,rm)
//   OBF:   0 = fp32 out, 1 = bf16 out.  Optional bias[N] and addC[M,N] (fp32).
//   WVM x WVN  : wave grid (must multiply to 8; blockDim = 256 = 8 waves)
//   ACM x ACN  : 16x16 accumulators per wave
// Block tile = (WVM*ACM*16) x (WVN*ACN*16) x 32.
// Requires M % BM == 0, N % BN == 0, K % 32 == 0.
// ---------------------------------------------------------------------------
template <int BMODE, int OBF, int WVM, int WVN, int ACM, int ACN>
__global__ __launch_bounds__(256) void gemm_wmma(
    const bf16_t* __restrict__ A, const bf16_t* __restrict__ Bm,
    const float* __restrict__ bias, const float* __restrict__ addC,
    void* __restrict__ outv, int M, int N, int K)
{
    static_assert(WVM * WVN == 8, "8 waves per workgroup");
    constexpr int BM = WVM * ACM * 16;
    constexpr int BN = WVN * ACN * 16;
    constexpr int BK = 32;
    constexpr int APITCH = BK + 8;                    // 80B rows (16B aligned)
    constexpr int BPITCH = BMODE ? (BN + 8) : (BK + 8);
    constexpr int ASZ = BM * APITCH;                  // bf16 elems per buffer
    constexpr int BSZ = BMODE ? (BK * BPITCH) : (BN * BPITCH);
    constexpr int ACPT = (BM * 4) / 256;              // A 16B chunks / thread
    constexpr int BCPT = (BN * 4) / 256;              // B 16B chunks / thread
    constexpr int BROWLEN = BMODE ? (BN / 8) : 4;     // 16B chunks per B row

    __shared__ bf16_t As[2 * ASZ];
    __shared__ bf16_t Bs[2 * BSZ];

    const int tid  = threadIdx.x;
    const int m0   = blockIdx.y * BM;
    const int n0   = blockIdx.x * BN;
    const int wave = tid >> 5;
    const int lane = tid & 31;
    const int wrow = wave / WVN;
    const int wcol = wave % WVN;
    const int ln   = lane & 15;
    const int kh   = lane >> 4;        // lane half (0/1)

    // ---- per-thread async copy slots (advance along K per tile)
    const bf16_t* aSrc[ACPT];
    unsigned int  aDst[ACPT];
    #pragma unroll
    for (int i = 0; i < ACPT; ++i) {
        const int c = tid + i * 256;
        const int row = c >> 2, seg = c & 3;
        aSrc[i] = A + (size_t)(m0 + row) * K + seg * 8;
        aDst[i] = lds_addr(&As[row * APITCH + seg * 8]);
    }
    const bf16_t* bSrc[BCPT];
    unsigned int  bDst[BCPT];
    #pragma unroll
    for (int i = 0; i < BCPT; ++i) {
        const int c = tid + i * 256;
        const int row = c / BROWLEN, seg = c % BROWLEN;
        if (BMODE == 0)
            bSrc[i] = Bm + (size_t)(n0 + row) * K + seg * 8;   // W row, K-seg
        else
            bSrc[i] = Bm + (size_t)row * N + n0 + seg * 8;     // K row, N-seg
        bDst[i] = lds_addr(&Bs[row * BPITCH + seg * 8]);
    }

    auto issue_tile = [&](int kt, int buf) {
        const unsigned int aoff = (unsigned int)buf * (ASZ * 2);   // bytes
        const unsigned int boff = (unsigned int)buf * (BSZ * 2);
        const size_t ka = (size_t)kt * BK;
        const size_t kb = BMODE ? (size_t)kt * BK * N : ka;
        #pragma unroll
        for (int i = 0; i < ACPT; ++i) async_ld_b128(aDst[i] + aoff, aSrc[i] + ka);
        #pragma unroll
        for (int i = 0; i < BCPT; ++i) async_ld_b128(bDst[i] + boff, bSrc[i] + kb);
    };

    v8f acc[ACM][ACN] = {};

    const int T = K / BK;
    issue_tile(0, 0);

    for (int kt = 0; kt < T; ++kt) {
        const int buf = kt & 1;
        // own copies for tile kt landed; barrier covers other waves' copies
        wait_asynccnt0();
        __syncthreads();
        // overlap: start filling the other buffer while computing this one
        if (kt + 1 < T) issue_tile(kt + 1, buf ^ 1);

        const bf16_t* asb = &As[buf * ASZ];
        const bf16_t* bsb = &Bs[buf * BSZ];

        // ---- build fragments, contiguous k order (k = 16*kh + s) for A and B
        FragBF afrag[ACM], bfrag[ACN];
        #pragma unroll
        for (int mi = 0; mi < ACM; ++mi) {
            const int mrow = wrow * (ACM * 16) + mi * 16 + ln;
            #pragma unroll
            for (int v = 0; v < 8; ++v) {
                const int k = kh * 16 + v * 2;
                afrag[mi].u[v] =
                    *(const unsigned int*)&asb[mrow * APITCH + k];
            }
        }
        #pragma unroll
        for (int ni = 0; ni < ACN; ++ni) {
            const int ncol = wcol * (ACN * 16) + ni * 16 + ln;
            if (BMODE == 0) {
                #pragma unroll
                for (int v = 0; v < 8; ++v) {
                    const int k = kh * 16 + v * 2;
                    bfrag[ni].u[v] =
                        *(const unsigned int*)&bsb[ncol * BPITCH + k];
                }
            } else {
                #pragma unroll
                for (int v = 0; v < 8; ++v) {
                    const int k = kh * 16 + v * 2;
                    bfrag[ni].h[2 * v]     = bsb[k * BPITCH + ncol];
                    bfrag[ni].h[2 * v + 1] = bsb[(k + 1) * BPITCH + ncol];
                }
            }
        }

        // ---- ACM*ACN WMMAs per wave per K-slab
        #pragma unroll
        for (int mi = 0; mi < ACM; ++mi)
            #pragma unroll
            for (int ni = 0; ni < ACN; ++ni)
                acc[mi][ni] = __builtin_amdgcn_wmma_f32_16x16x32_bf16(
                    false, afrag[mi].v, false, bfrag[ni].v,
                    (short)0, acc[mi][ni], false, false);

        // all waves done reading this buffer before it is refilled at kt+2
        __syncthreads();
    }

    // ---- epilogue: C layout = VGPR r -> M = kh*8 + r, N = ln
    #pragma unroll
    for (int mi = 0; mi < ACM; ++mi) {
        #pragma unroll
        for (int ni = 0; ni < ACN; ++ni) {
            const int mbase = m0 + wrow * (ACM * 16) + mi * 16 + kh * 8;
            const int ncol  = n0 + wcol * (ACN * 16) + ni * 16 + ln;
            const float badd = bias ? bias[ncol] : 0.0f;
            #pragma unroll
            for (int r = 0; r < 8; ++r) {
                const size_t idx = (size_t)(mbase + r) * N + ncol;
                float val = acc[mi][ni][r] + badd;
                if (addC) val += addC[idx];
                if (OBF)  ((bf16_t*)outv)[idx] = (bf16_t)val;
                else      ((float*)outv)[idx]  = val;
            }
        }
    }
}

// ---------------------------------------------------------------------------
// Elementwise helpers
// ---------------------------------------------------------------------------
__global__ void f32_to_bf16_kernel(const float* __restrict__ in,
                                   bf16_t* __restrict__ out, size_t n)
{
    for (size_t i = (size_t)blockIdx.x * blockDim.x + threadIdx.x; i < n;
         i += (size_t)gridDim.x * blockDim.x)
        out[i] = (bf16_t)in[i];
}

__global__ void add_bias_kernel(const float* __restrict__ a,
                                const float* __restrict__ b,
                                float* __restrict__ o, int n)
{
    int i = blockIdx.x * blockDim.x + threadIdx.x;
    if (i < n) o[i] = a[i] + b[i];
}

// Fused LSTM cell (PyTorch gate order i,f,g,o). Gates are fp32 pre-activations
// [B, 4H]; c is fp32 state; h written as bf16 (recurrent GEMM + attention feed).
__global__ void lstm_cell_kernel(const float* __restrict__ gates,
                                 float* __restrict__ c,
                                 bf16_t* __restrict__ h_bf,
                                 bf16_t* __restrict__ hs_t, int H)
{
    const int idx = blockIdx.x * blockDim.x + threadIdx.x;   // 0..B*H-1
    const int b = idx / H;
    const int j = idx - b * H;
    const float* g = gates + (size_t)b * 4 * H;
    const float gi = 1.0f / (1.0f + expf(-g[j]));
    const float gf = 1.0f / (1.0f + expf(-g[H + j]));
    const float gg = tanhf(g[2 * H + j]);
    const float go = 1.0f / (1.0f + expf(-g[3 * H + j]));
    const float cn = gf * c[idx] + gi * gg;
    c[idx] = cn;
    const float h = go * tanhf(cn);
    h_bf[idx] = (bf16_t)h;
    hs_t[idx] = (bf16_t)h;
}

// Row softmax of a 128x128 matrix, bf16 output. blockDim.x == ncols == 128.
__global__ void softmax_rows_kernel(const float* __restrict__ W,
                                    bf16_t* __restrict__ A, int ncols)
{
    __shared__ float red[128];
    const int r = blockIdx.x, t = threadIdx.x;
    const float x = W[(size_t)r * ncols + t];
    red[t] = x;
    __syncthreads();
    for (int s = 64; s > 0; s >>= 1) {
        if (t < s) red[t] = fmaxf(red[t], red[t + s]);
        __syncthreads();
    }
    const float m = red[0];
    __syncthreads();
    const float e = expf(x - m);
    red[t] = e;
    __syncthreads();
    for (int s = 64; s > 0; s >>= 1) {
        if (t < s) red[t] += red[t + s];
        __syncthreads();
    }
    A[(size_t)r * ncols + t] = (bf16_t)(e / red[0]);
}

// ---------------------------------------------------------------------------
// Host orchestration
// ---------------------------------------------------------------------------
extern "C" void kernel_launch(void* const* d_in, const int* in_sizes, int n_in,
                              void* d_out, int out_size, void* d_ws, size_t ws_size,
                              hipStream_t stream)
{
    constexpr int S = 128, Bn = 256, IN = 1024, H = 1024, OUT = 1024;
    constexpr int G = 4 * H;                       // 4096 gate columns
    (void)in_sizes; (void)n_in; (void)out_size; (void)ws_size;

    const float* inputs = (const float*)d_in[0];   // [S,B,IN]
    const float* h0     = (const float*)d_in[1];   // [B,H]
    const float* c0     = (const float*)d_in[2];   // [B,H]
    const float* W_ih   = (const float*)d_in[3];   // [4H,IN]
    const float* W_hh   = (const float*)d_in[4];   // [4H,H]
    const float* b_ih   = (const float*)d_in[5];   // [4H]
    const float* b_hh   = (const float*)d_in[6];   // [4H]
    const float* attnW  = (const float*)d_in[7];   // [S,S]
    const float* linW   = (const float*)d_in[8];   // [OUT,H]
    const float* linb   = (const float*)d_in[9];   // [OUT]
    float* out = (float*)d_out;                    // [S*B, OUT] fp32

    // Workspace bump allocator (256B aligned)
    char* base = (char*)d_ws;
    size_t off = 0;
    auto alloc = [&](size_t bytes) -> void* {
        void* p = base + off;
        off = (off + bytes + 255) & ~(size_t)255;
        return p;
    };

    bf16_t* xin_bf   = (bf16_t*)alloc((size_t)S * Bn * IN * 2);   // 67 MB
    bf16_t* wih_bf   = (bf16_t*)alloc((size_t)G * IN * 2);        // 8 MB
    bf16_t* whh_bf   = (bf16_t*)alloc((size_t)G * H * 2);         // 8 MB
    bf16_t* lin_bf   = (bf16_t*)alloc((size_t)OUT * H * 2);       // 2 MB
    bf16_t* asoft_bf = (bf16_t*)alloc((size_t)S * S * 2);
    float*  bsum     = (float*)alloc((size_t)G * 4);
    float*  gates_pre= (float*)alloc((size_t)S * Bn * G * 4);     // 512 MB
    float*  gates_cur= (float*)alloc((size_t)Bn * G * 4);         // 4 MB
    bf16_t* h_bf     = (bf16_t*)alloc((size_t)Bn * H * 2);
    float*  c_buf    = (float*)alloc((size_t)Bn * H * 4);
    bf16_t* hs_bf    = (bf16_t*)alloc((size_t)S * Bn * H * 2);    // 67 MB
    bf16_t* att_bf   = (bf16_t*)alloc((size_t)S * Bn * H * 2);    // 67 MB

    // ---- precision staging: fp32 -> bf16 (weights, inputs, h0); c stays fp32
    hipMemcpyAsync(c_buf, c0, (size_t)Bn * H * 4,
                   hipMemcpyDeviceToDevice, stream);
    f32_to_bf16_kernel<<<4096, 256, 0, stream>>>(inputs, xin_bf, (size_t)S * Bn * IN);
    f32_to_bf16_kernel<<<2048, 256, 0, stream>>>(W_ih, wih_bf, (size_t)G * IN);
    f32_to_bf16_kernel<<<2048, 256, 0, stream>>>(W_hh, whh_bf, (size_t)G * H);
    f32_to_bf16_kernel<<<1024, 256, 0, stream>>>(linW, lin_bf, (size_t)OUT * H);
    f32_to_bf16_kernel<<<256, 256, 0, stream>>>(h0, h_bf, (size_t)Bn * H);
    add_bias_kernel<<<G / 256, 256, 0, stream>>>(b_ih, b_hh, bsum, G);

    // ---- hoisted input projection: gates_pre = X @ W_ih^T + (b_ih + b_hh)
    //      throughput config: 128x256 tile, 4x4 accumulators per wave
    gemm_wmma<0, 0, 2, 4, 4, 4><<<dim3(G / 256, (S * Bn) / 128), 256, 0, stream>>>(
        xin_bf, wih_bf, bsum, nullptr, gates_pre, S * Bn, G, IN);

    // ---- sequential recurrence: 128 x (GEMM + fused cell)
    //      latency config: 64x128 tile keeps 128 workgroups per step
    for (int t = 0; t < S; ++t) {
        gemm_wmma<0, 0, 2, 4, 2, 2><<<dim3(G / 128, Bn / 64), 256, 0, stream>>>(
            h_bf, whh_bf, nullptr, gates_pre + (size_t)t * Bn * G,
            gates_cur, Bn, G, H);
        lstm_cell_kernel<<<(Bn * H) / 256, 256, 0, stream>>>(
            gates_cur, c_buf, h_bf, hs_bf + (size_t)t * Bn * H, H);
    }

    // ---- attention: A = softmax(attnW) rows; att = A @ hs  (K-major B GEMM)
    softmax_rows_kernel<<<S, S, 0, stream>>>(attnW, asoft_bf, S);
    gemm_wmma<1, 1, 2, 4, 4, 4><<<dim3((Bn * H) / 256, S / 128), 256, 0, stream>>>(
        asoft_bf, hs_bf, nullptr, nullptr, att_bf, S, Bn * H, S);

    // ---- output projection: out = att @ linW^T + linb   (fp32 out)
    gemm_wmma<0, 0, 2, 4, 4, 4><<<dim3(OUT / 256, (S * Bn) / 128), 256, 0, stream>>>(
        att_bf, lin_bf, linb, nullptr, out, S * Bn, OUT, H);
}
